// MultiHeadAttention_6940667150871
// MI455X (gfx1250) — matrix operand order repaired and checked by reading, not compiled, not run on hardware
//
#include <hip/hip_runtime.h>
#include <hip/hip_bf16.h>

// ---------------------------------------------------------------------------
// MHA block for MI455X (gfx1250, wave32, WMMA bf16 16x16x32, f32 accumulate)
// ---------------------------------------------------------------------------

#define D_MODEL 1024
#define N_HEADS 16
#define D_K     64
#define SEQ     2048
#define BATCH   2
#define TOKENS  (BATCH * SEQ)   // 4096

typedef __attribute__((ext_vector_type(16))) __bf16 bf16x16;
typedef __attribute__((ext_vector_type(8)))  __bf16 bf16x8;
typedef __attribute__((ext_vector_type(8)))  float  floatx8;

// ---- async global->LDS path (gfx1250), guarded so fallback always compiles
#if defined(__AMDGCN__) && \
    __has_builtin(__builtin_amdgcn_global_load_async_to_lds_b128) && \
    __has_builtin(__builtin_amdgcn_s_wait_asynccnt)
#define USE_ASYNC 1
#else
#define USE_ASYNC 0
#endif

#if USE_ASYNC
typedef int v4i_ __attribute__((vector_size(16)));
static __device__ __forceinline__ void async_ld128(const void* g, void* l) {
    // generic->as(1): same 64-bit value; generic->as(3): low 32 bits are the
    // LDS offset (aperture rule), as(3) pointers are 32-bit.
    __builtin_amdgcn_global_load_async_to_lds_b128(
        (__attribute__((address_space(1))) v4i_*)(unsigned long long)(uintptr_t)g,
        (__attribute__((address_space(3))) v4i_*)(unsigned)(uintptr_t)l,
        0, 0);
}
#define WAIT_ASYNC(n) __builtin_amdgcn_s_wait_asynccnt(n)
#else
#define WAIT_ASYNC(n)
#endif

static __device__ __forceinline__ __bf16 f2bf(float f) {
    unsigned u = __builtin_bit_cast(unsigned, f);
    unsigned r = u + 0x7FFFu + ((u >> 16) & 1u);   // round-to-nearest-even
    unsigned short h = (unsigned short)(r >> 16);
    return __builtin_bit_cast(__bf16, h);
}

static __device__ __forceinline__ bf16x16 joinv(bf16x8 lo, bf16x8 hi) {
    union { bf16x16 v; bf16x8 h[2]; } u;
    u.h[0] = lo; u.h[1] = hi;
    return u.v;
}

static __device__ __forceinline__ floatx8 wmma_bf16(bf16x16 a, bf16x16 b, floatx8 c) {
    return __builtin_amdgcn_wmma_f32_16x16x32_bf16(false, a, false, b,
                                                   (short)0, c, false, false);
}

// ---------------------------------------------------------------------------
// 1) LayerNorm: x [4096,1024] f32 -> xn bf16
// ---------------------------------------------------------------------------
__global__ __launch_bounds__(256)
void ln_kernel(const float* __restrict__ x, const float* __restrict__ gamma,
               const float* __restrict__ beta, __bf16* __restrict__ xn) {
    const int t   = blockIdx.x;
    const int tid = threadIdx.x;
    const float* row = x + (size_t)t * D_MODEL;

    float s = 0.f, ss = 0.f;
    float vals[4];
#pragma unroll
    for (int i = 0; i < 4; ++i) {
        float v = row[tid + i * 256];
        vals[i] = v; s += v; ss += v * v;
    }
#pragma unroll
    for (int off = 1; off < 32; off <<= 1) {
        s  += __shfl_xor(s,  off, 32);
        ss += __shfl_xor(ss, off, 32);
    }
    __shared__ float sh[16];
    const int wave = tid >> 5, lane = tid & 31;
    if (lane == 0) { sh[wave] = s; sh[8 + wave] = ss; }
    __syncthreads();
    s = 0.f; ss = 0.f;
#pragma unroll
    for (int w = 0; w < 8; ++w) { s += sh[w]; ss += sh[8 + w]; }

    const float mu   = s * (1.0f / D_MODEL);
    const float var  = ss * (1.0f / D_MODEL) - mu * mu;
    const float rstd = rsqrtf(var + 1e-5f);
#pragma unroll
    for (int i = 0; i < 4; ++i) {
        int c = tid + i * 256;
        xn[(size_t)t * D_MODEL + c] = f2bf((vals[i] - mu) * rstd * gamma[c] + beta[c]);
    }
}

// ---------------------------------------------------------------------------
// 2) Weight f32 -> bf16 (layout unchanged: [out,in] row-major)
// ---------------------------------------------------------------------------
__global__ __launch_bounds__(256)
void cvt_kernel(const float* __restrict__ w, __bf16* __restrict__ o, int n) {
    int i = blockIdx.x * blockDim.x + threadIdx.x;
    if (i < n) o[i] = f2bf(w[i]);
}

// ---------------------------------------------------------------------------
// 3) Fused QKV projection: y = Xn @ W^T  (M=4096, N=1024, K=1024) x3 weights
//    Block = 256 thr (8 waves), block tile 64(M) x 256(N); wave tile 32x64:
//    2 A-frags x 4 B-frags -> 8 WMMA per K-step, B reused across both M-frags.
//    grid.z in {0,1,2} selects (Wq->Q, Wk->K, Wv->V); head-interleaved output.
// ---------------------------------------------------------------------------
__global__ __launch_bounds__(256)
void qkv_kernel(const __bf16* __restrict__ xn,
                const __bf16* __restrict__ wqb, const __bf16* __restrict__ wkb,
                const __bf16* __restrict__ wvb,
                __bf16* __restrict__ Qb, __bf16* __restrict__ Kb,
                __bf16* __restrict__ Vb) {
    const __bf16* w;
    __bf16* out;
    if (blockIdx.z == 0)      { w = wqb; out = Qb; }
    else if (blockIdx.z == 1) { w = wkb; out = Kb; }
    else                      { w = wvb; out = Vb; }

    const int tid  = threadIdx.x;
    const int wave = tid >> 5;
    const int lane = tid & 31;
    const int l16  = lane & 15;
    const bool hh  = lane >= 16;
    const int wm   = wave >> 2;                        // 0..1
    const int wn   = wave & 3;                         // 0..3
    const int m0   = blockIdx.y * 64 + wm * 32;
    const int n0   = blockIdx.x * 256 + wn * 64;

    floatx8 acc[2][4] = {};
    const __bf16* arow0 = xn + (size_t)(m0 + l16) * D_MODEL;
    const __bf16* arow1 = xn + (size_t)(m0 + 16 + l16) * D_MODEL;

    for (int k = 0; k < D_MODEL; k += 32) {
        bf16x16 a0 = joinv(*(const bf16x8*)(arow0 + k + (hh ? 8 : 0)),
                           *(const bf16x8*)(arow0 + k + (hh ? 24 : 16)));
        bf16x16 a1 = joinv(*(const bf16x8*)(arow1 + k + (hh ? 8 : 0)),
                           *(const bf16x8*)(arow1 + k + (hh ? 24 : 16)));
#pragma unroll
        for (int t = 0; t < 4; ++t) {
            const __bf16* brow =
                w + (size_t)(n0 + t * 16 + l16) * D_MODEL + k + (hh ? 8 : 0);
            bf16x16 bfrag = joinv(*(const bf16x8*)brow, *(const bf16x8*)(brow + 16));
            acc[0][t] = wmma_bf16(a0, bfrag, acc[0][t]);
            acc[1][t] = wmma_bf16(a1, bfrag, acc[1][t]);
        }
    }

#pragma unroll
    for (int mi = 0; mi < 2; ++mi)
#pragma unroll
        for (int t = 0; t < 4; ++t) {
            const int o = n0 + t * 16 + l16;
            const int h = o >> 6, d = o & 63;
#pragma unroll
            for (int r = 0; r < 8; ++r) {
                const int tok = m0 + mi * 16 + r + (hh ? 8 : 0);
                const int bb  = tok >> 11, sq = tok & (SEQ - 1);
                out[(((size_t)(bb * N_HEADS + h)) * SEQ + sq) * D_K + d] =
                    f2bf(acc[mi][t][r]);
            }
        }
}

// ---------------------------------------------------------------------------
// 4) Flash attention: per (b,h), 64-query block tiles x stream of 64-key tiles
//    Block = 128 thr (4 waves), each wave owns 16 query rows.
//    Double-buffered K/V staging; K tile via async global->LDS when available.
// ---------------------------------------------------------------------------
__global__ __launch_bounds__(128)
void attn_kernel(const __bf16* __restrict__ Q, const __bf16* __restrict__ K,
                 const __bf16* __restrict__ V, __bf16* __restrict__ att) {
    __shared__ __attribute__((aligned(32))) __bf16 kt[2][64][64];    // [buf][key][d]
    __shared__ __attribute__((aligned(32))) __bf16 vt[2][64][64];    // [buf][d][key]
    __shared__ __attribute__((aligned(32))) __bf16 pt[4][16][64];    // per-wave P

    const int tid  = threadIdx.x;
    const int wave = tid >> 5;
    const int lane = tid & 31;
    const int l16  = lane & 15;
    const bool hh  = lane >= 16;
    const int bh   = blockIdx.y;                    // 0..31
    const int q0   = blockIdx.x * 64 + wave * 16;   // query row base for wave
    const size_t base = (size_t)bh * SEQ * D_K;

    // Q A-fragments (K-dim = d = 64 -> 2 fragments), loaded once
    bf16x16 aq[2];
    {
        const __bf16* qrow = Q + base + (size_t)(q0 + l16) * D_K;
#pragma unroll
        for (int f = 0; f < 2; ++f) {
            const int kk = f * 32;
            aq[f] = joinv(*(const bf16x8*)(qrow + kk + (hh ? 8 : 0)),
                          *(const bf16x8*)(qrow + kk + (hh ? 24 : 16)));
        }
    }

    floatx8 o[4] = {};
    float m[8], l[8];
#pragma unroll
    for (int r = 0; r < 8; ++r) { m[r] = -1e30f; l[r] = 0.f; }

    // ---- staging helper (lambda): tile jt -> buffer nb ----
    auto stage = [&](int jt, int nb) {
        // K tile [64 keys][64 d], row-major, per-wave 16-row slab
#if USE_ASYNC
        {
            const __bf16* kg = K + base + (size_t)(jt * 64 + wave * 16) * D_K;
            __bf16* kl = &kt[nb][wave * 16][0];
            const int row = lane >> 3;             // 0..3
            const int c0  = (lane & 7) * 8;        // 16B chunk
#pragma unroll
            for (int i = 0; i < 4; ++i)
                async_ld128(kg + (size_t)(i * 4 + row) * D_K + c0,
                            kl + (i * 4 + row) * 64 + c0);
        }
#else
        {
            const __bf16* kg = K + base + (size_t)(jt * 64) * D_K;
            for (int c = tid; c < 64 * 4; c += 128) {
                const int key = c >> 2, d0 = (c & 3) * 16;
                *(bf16x16*)&kt[nb][key][d0] = *(const bf16x16*)(kg + key * D_K + d0);
            }
        }
#endif
        // V tile transposed to [d][key]; per-wave 16-row slab, VALU transpose
        {
            const int vrow = wave * 16 + (lane >> 1);
            const int vcol = (lane & 1) * 32;
            const __bf16* vg = V + base + (size_t)(jt * 64 + vrow) * D_K + vcol;
#pragma unroll
            for (int i = 0; i < 4; ++i) {
                bf16x8 v = *(const bf16x8*)(vg + i * 8);
#pragma unroll
                for (int e = 0; e < 8; ++e) vt[nb][vcol + i * 8 + e][vrow] = v[e];
            }
        }
    };

    const int NT = SEQ / 64;        // 32 key tiles
    stage(0, 0);                    // prologue

    for (int j = 0; j < NT; ++j) {
        const int cb = j & 1;
        if (j + 1 < NT) {
            stage(j + 1, cb ^ 1);   // overlaps with compute of tile j
            WAIT_ASYNC(4);          // tile j's async K loads complete (in-order)
        } else {
            WAIT_ASYNC(0);
        }
        __syncthreads();            // tile j fully staged across all waves

        // ---- scores S = Q K^T * 1/sqrt(64) : 4 N-tiles x 2 WMMA ----
        floatx8 s[4];
#pragma unroll
        for (int nt = 0; nt < 4; ++nt) {
            const int n = nt * 16 + l16;
            bf16x16 b0 = joinv(*(const bf16x8*)&kt[cb][n][hh ? 8 : 0], *(const bf16x8*)&kt[cb][n][hh ? 24 : 16]);
            bf16x16 b1 = joinv(*(const bf16x8*)&kt[cb][n][hh ? 40 : 32], *(const bf16x8*)&kt[cb][n][hh ? 56 : 48]);
            floatx8 c = {};
            c = wmma_bf16(aq[0], b0, c);
            c = wmma_bf16(aq[1], b1, c);
            s[nt] = c * 0.125f;
        }

        // ---- online softmax (rows per-VGPR; reduce across 16-lane half) ----
        float mnew[8], alpha[8], rs[8];
#pragma unroll
        for (int r = 0; r < 8; ++r) {
            float v = fmaxf(fmaxf(s[0][r], s[1][r]), fmaxf(s[2][r], s[3][r]));
#pragma unroll
            for (int off = 1; off < 16; off <<= 1) v = fmaxf(v, __shfl_xor(v, off, 32));
            mnew[r]  = fmaxf(m[r], v);
            alpha[r] = __expf(m[r] - mnew[r]);
            rs[r]    = 0.f;
        }
#pragma unroll
        for (int nt = 0; nt < 4; ++nt)
#pragma unroll
            for (int r = 0; r < 8; ++r) {
                float p = __expf(s[nt][r] - mnew[r]);
                s[nt][r] = p;
                rs[r]   += p;
            }
#pragma unroll
        for (int r = 0; r < 8; ++r) {
            float v = rs[r];
#pragma unroll
            for (int off = 1; off < 16; off <<= 1) v += __shfl_xor(v, off, 32);
            l[r] = l[r] * alpha[r] + v;
            m[r] = mnew[r];
        }
#pragma unroll
        for (int dt = 0; dt < 4; ++dt)
#pragma unroll
            for (int r = 0; r < 8; ++r) o[dt][r] *= alpha[r];

        // ---- P (C-layout) -> wave-private LDS (row-major bf16) ----
#pragma unroll
        for (int nt = 0; nt < 4; ++nt)
#pragma unroll
            for (int r = 0; r < 8; ++r)
                pt[wave][r + (hh ? 8 : 0)][nt * 16 + l16] = f2bf(s[nt][r]);

        // ---- O += P @ V : A-fragments of P from LDS, B from transposed V ----
        const __bf16* prow = &pt[wave][l16][0];
#pragma unroll
        for (int f = 0; f < 2; ++f) {
            const int kk = f * 32;
            bf16x16 ap = joinv(*(const bf16x8*)(prow + kk + (hh ? 8 : 0)),
                               *(const bf16x8*)(prow + kk + (hh ? 24 : 16)));
#pragma unroll
            for (int dt = 0; dt < 4; ++dt) {
                const int d = dt * 16 + l16;
                bf16x16 bv = joinv(*(const bf16x8*)&vt[cb][d][kk + (hh ? 8 : 0)],
                                   *(const bf16x8*)&vt[cb][d][kk + (hh ? 24 : 16)]);
                o[dt] = wmma_bf16(ap, bv, o[dt]);
            }
        }
        __syncthreads();            // buffer cb free for reuse at iter j+2
    }

    // ---- finalize: O / l, store attended [tok][h*64+d] bf16 ----
    const int b = bh >> 4, h = bh & 15;
#pragma unroll
    for (int dt = 0; dt < 4; ++dt)
#pragma unroll
        for (int r = 0; r < 8; ++r) {
            const int tok = b * SEQ + q0 + r + (hh ? 8 : 0);
            const float inv = 1.0f / l[r];
            att[(size_t)tok * D_MODEL + h * D_K + dt * 16 + l16] = f2bf(o[dt][r] * inv);
        }
}

// ---------------------------------------------------------------------------
// 5) Output projection + bias + residual: out = att @ Wo^T + b_o + x   (f32)
//    Same 32x64 wave tile as qkv_kernel.
// ---------------------------------------------------------------------------
__global__ __launch_bounds__(256)
void outproj_kernel(const __bf16* __restrict__ att, const __bf16* __restrict__ w,
                    const float* __restrict__ bias, const float* __restrict__ x,
                    float* __restrict__ out) {
    const int tid  = threadIdx.x;
    const int wave = tid >> 5;
    const int lane = tid & 31;
    const int l16  = lane & 15;
    const bool hh  = lane >= 16;
    const int wm   = wave >> 2;
    const int wn   = wave & 3;
    const int m0   = blockIdx.y * 64 + wm * 32;
    const int n0   = blockIdx.x * 256 + wn * 64;

    floatx8 acc[2][4] = {};
    const __bf16* arow0 = att + (size_t)(m0 + l16) * D_MODEL;
    const __bf16* arow1 = att + (size_t)(m0 + 16 + l16) * D_MODEL;

    for (int k = 0; k < D_MODEL; k += 32) {
        bf16x16 a0 = joinv(*(const bf16x8*)(arow0 + k + (hh ? 8 : 0)),
                           *(const bf16x8*)(arow0 + k + (hh ? 24 : 16)));
        bf16x16 a1 = joinv(*(const bf16x8*)(arow1 + k + (hh ? 8 : 0)),
                           *(const bf16x8*)(arow1 + k + (hh ? 24 : 16)));
#pragma unroll
        for (int t = 0; t < 4; ++t) {
            const __bf16* brow =
                w + (size_t)(n0 + t * 16 + l16) * D_MODEL + k + (hh ? 8 : 0);
            bf16x16 bfrag = joinv(*(const bf16x8*)brow, *(const bf16x8*)(brow + 16));
            acc[0][t] = wmma_bf16(a0, bfrag, acc[0][t]);
            acc[1][t] = wmma_bf16(a1, bfrag, acc[1][t]);
        }
    }

#pragma unroll
    for (int mi = 0; mi < 2; ++mi)
#pragma unroll
        for (int t = 0; t < 4; ++t) {
            const int o = n0 + t * 16 + l16;
#pragma unroll
            for (int r = 0; r < 8; ++r) {
                const int tok = m0 + mi * 16 + r + (hh ? 8 : 0);
                out[(size_t)tok * D_MODEL + o] =
                    acc[mi][t][r] + bias[o] + x[(size_t)tok * D_MODEL + o];
            }
        }
}

// ---------------------------------------------------------------------------
// Launch
// ---------------------------------------------------------------------------
extern "C" void kernel_launch(void* const* d_in, const int* in_sizes, int n_in,
                              void* d_out, int out_size, void* d_ws, size_t ws_size,
                              hipStream_t stream) {
    (void)in_sizes; (void)n_in; (void)out_size; (void)ws_size;
    const float* x     = (const float*)d_in[0];
    const float* wq    = (const float*)d_in[1];
    const float* wk    = (const float*)d_in[2];
    const float* wv    = (const float*)d_in[3];
    const float* wo    = (const float*)d_in[4];
    const float* bo    = (const float*)d_in[5];
    const float* gamma = (const float*)d_in[6];
    const float* beta  = (const float*)d_in[7];

    char* ws = (char*)d_ws;
    const size_t MB = (size_t)1 << 20;
    __bf16* xn   = (__bf16*)(ws + 0 * MB);    // 8 MB  (4096x1024 bf16)
    __bf16* wqb  = (__bf16*)(ws + 8 * MB);    // 2 MB
    __bf16* wkb  = (__bf16*)(ws + 10 * MB);   // 2 MB
    __bf16* wvb  = (__bf16*)(ws + 12 * MB);   // 2 MB
    __bf16* wob  = (__bf16*)(ws + 14 * MB);   // 2 MB
    __bf16* Qb   = (__bf16*)(ws + 16 * MB);   // 8 MB  [B*H][S][64]
    __bf16* Kb   = (__bf16*)(ws + 24 * MB);   // 8 MB
    __bf16* Vb   = (__bf16*)(ws + 32 * MB);   // 8 MB
    __bf16* attb = (__bf16*)(ws + 40 * MB);   // 8 MB  [tok][1024]

    ln_kernel<<<TOKENS, 256, 0, stream>>>(x, gamma, beta, xn);

    const int NW = D_MODEL * D_MODEL;
    cvt_kernel<<<4096, 256, 0, stream>>>(wq, wqb, NW);
    cvt_kernel<<<4096, 256, 0, stream>>>(wk, wkb, NW);
    cvt_kernel<<<4096, 256, 0, stream>>>(wv, wvb, NW);
    cvt_kernel<<<4096, 256, 0, stream>>>(wo, wob, NW);

    qkv_kernel<<<dim3(D_MODEL / 256, TOKENS / 64, 3), 256, 0, stream>>>(
        xn, wqb, wkb, wvb, Qb, Kb, Vb);

    attn_kernel<<<dim3(SEQ / 64, BATCH * N_HEADS), 128, 0, stream>>>(Qb, Kb, Vb, attb);

    outproj_kernel<<<dim3(D_MODEL / 256, TOKENS / 64), 256, 0, stream>>>(
        attb, wob, bo, x, (float*)d_out);
}
